// HypGRU_35321811042418
// MI455X (gfx1250) — compile-verified, matrix operand order checked by reading, not run on hardware
//
#include <hip/hip_runtime.h>
#include <math.h>

typedef __attribute__((ext_vector_type(16))) _Float16 v16h;
typedef __attribute__((ext_vector_type(8)))  _Float16 v8h;
typedef __attribute__((ext_vector_type(8)))  float    v8f;

#define DH   128
#define DT   100
#define DIN  128
#define KX   256   // DIN+DT=228 padded to 256 (8 WMMA k-steps of 32)
#define GIC  384   // Gi columns: [W_ir | W_ih | W_iz]
#define GHC  256   // Gh columns: [W_hr | W_hz]
#define SGW  648   // LDS stage row stride: 384 + 256 + 8 pad floats
#define BR   32    // rows per block (two 16-row WMMA halves)

// ---------------- wave32 helpers ----------------

__device__ __forceinline__ float wsum(float v) {
#pragma unroll
  for (int o = 16; o > 0; o >>= 1) v += __shfl_xor(v, o, 32);
  return v;
}

__device__ __forceinline__ float artanh_c(float x) {
  x = fminf(fmaxf(x, -1.0f + 1e-5f), 1.0f - 1e-5f);
  return 0.5f * logf((1.0f + x) / (1.0f - x));
}

__device__ __forceinline__ void ld4(const float* p, float* o) {
  float4 v = *(const float4*)p;
  o[0] = v.x; o[1] = v.y; o[2] = v.z; o[3] = v.w;
}
__device__ __forceinline__ void st4(float* p, const float* o) {
  *(float4*)p = make_float4(o[0], o[1], o[2], o[3]);
}

struct alignas(8) h4 { _Float16 v[4]; };
__device__ __forceinline__ void sth4(_Float16* p, const float* o) {
  h4 t;
#pragma unroll
  for (int j = 0; j < 4; ++j) t.v[j] = (_Float16)o[j];
  *(h4*)p = t;
}

// norm over 128 dims (4 per lane), clamped like reference _norm()
__device__ __forceinline__ float norm4(const float* v) {
  float s = 0.f;
#pragma unroll
  for (int j = 0; j < 4; ++j) s += v[j] * v[j];
  return fmaxf(sqrtf(wsum(s)), 1e-15f);
}

// Poincare-ball Mobius addition (c = 1), 4 channels per lane
__device__ __forceinline__ void mobius_add4(const float* x, const float* y, float* o) {
  float sx = 0.f, sy = 0.f, sxy = 0.f;
#pragma unroll
  for (int j = 0; j < 4; ++j) { sx += x[j] * x[j]; sy += y[j] * y[j]; sxy += x[j] * y[j]; }
  float x2 = wsum(sx), y2 = wsum(sy), xy = wsum(sxy);
  float a   = 1.f + 2.f * xy + y2;
  float b   = 1.f - x2;
  float den = fmaxf(1.f + 2.f * xy + x2 * y2, 1e-15f);
  float inv = 1.f / den;
#pragma unroll
  for (int j = 0; j < 4; ++j) o[j] = (a * x[j] + b * y[j]) * inv;
}

// shared core of mobius_matvec / mobius_pointwise_mul
__device__ __forceinline__ void mscale4(const float* mx, float xnorm, float* o) {
  float mxn = norm4(mx);
  float xn  = fmaxf(xnorm, 1e-15f);
  float t   = tanhf(mxn / xn * artanh_c(xn));
  float s   = (mxn <= 1e-10f) ? 0.f : (t / mxn);
#pragma unroll
  for (int j = 0; j < 4; ++j) o[j] = s * mx[j];
}

__device__ __forceinline__ void logmap0_4(const float* x, float* o) {
  float n = norm4(x);
  float s = artanh_c(n) / n;
#pragma unroll
  for (int j = 0; j < 4; ++j) o[j] = s * x[j];
}

// ---------------- WMMA fragment load (16x32 f16, ISA lane layout) ----------------
// lanes 0-15: M=0..15, K 0..7 / 16..23 ; lanes 16-31: same M, K 8..15 / 24..31

union FragU { v16h v; v8h h[2]; };

__device__ __forceinline__ v16h frag_ld(const _Float16* base, int ld, int k) {
  int lane = threadIdx.x & 31;
  const _Float16* p = base + (size_t)(lane & 15) * ld + k + ((lane >> 4) << 3);
  FragU u;
  u.h[0] = *(const v8h*)p;
  u.h[1] = *(const v8h*)(p + 16);
  return u.v;
}

// strided accumulator -> LDS stage store; each of the 8 rows gets a distinct
// constant ds offset (i*SGW floats)
__device__ __forceinline__ void stacc(float* sGbase, int rowHalf, int col, v8f acc) {
  int lane = threadIdx.x & 31;
  int mb = ((lane >> 4) << 3) + rowHalf;           // M = i + 8*(lane>=16) (+ half*16)
  float* p = sGbase + (size_t)mb * SGW + col + (lane & 15);
#pragma unroll
  for (int i = 0; i < 8; ++i) p[(size_t)i * SGW] = acc[i];
}

// ---------------- weight / bias prep (tiny, runs once per launch) ----------------

__global__ void prep_weights(const float* __restrict__ wih, const float* __restrict__ whh,
                             _Float16* __restrict__ Wi16, _Float16* __restrict__ Whrz16,
                             _Float16* __restrict__ Whh16) {
  int which = blockIdx.y;
  int idx = blockIdx.x * blockDim.x + threadIdx.x;
  if (which == 0) {                       // Wi16: [384][KX], K zero-padded
    if (idx >= GIC * KX) return;
    int m = idx / KX, k = idx - m * KX;
    Wi16[idx] = (k < DIN + DT) ? (_Float16)wih[m * (DIN + DT) + k] : (_Float16)0.f;
  } else if (which == 1) {                // Whrz16: rows [W_hr ; W_hz], [256][128]
    if (idx >= GHC * DH) return;
    int j = idx / DH, k = idx - j * DH;
    int src = (j < DH) ? j : (2 * DH + (j - DH));
    Whrz16[idx] = (_Float16)whh[src * DH + k];
  } else {                                // Whh16: W_hh rows, [128][128]
    if (idx >= DH * DH) return;
    int j = idx / DH, k = idx - j * DH;
    Whh16[idx] = (_Float16)whh[(DH + j) * DH + k];
  }
}

__global__ void prep_bias(const float* __restrict__ bias, float* __restrict__ b_hyp) {
  int row = blockIdx.x;         // 3 rows, one wave each
  int lane = threadIdx.x;
  float v[4];
  ld4(bias + row * DH + 4 * lane, v);
  float s = 0.f;
#pragma unroll
  for (int j = 0; j < 4; ++j) s += v[j] * v[j];
  float un = fmaxf(sqrtf(wsum(s)), 1e-15f);
  float sc = tanhf(un) / un;    // expmap0
#pragma unroll
  for (int j = 0; j < 4; ++j) v[j] *= sc;
  st4(b_hyp + row * DH + 4 * lane, v);
}

// ------- fully fused HypGRU: 32 rows per block, 8 wave32s, dual-acc B reuse -------

__global__ void __launch_bounds__(256)
hypgru_fused(const float* __restrict__ mem_input, const float* __restrict__ mem,
             const float* __restrict__ ts, const float* __restrict__ mem_ts,
             const float* __restrict__ hin,
             const float* __restrict__ time_w, const float* __restrict__ time_b,
             const _Float16* __restrict__ Wi16, const _Float16* __restrict__ Whrz16,
             const _Float16* __restrict__ Whh16, const float* __restrict__ b_hyp,
             float* __restrict__ out, int N) {
  __shared__ alignas(16) _Float16 sXA[BR][KX];   // x rows, f16, K padded
  __shared__ alignas(16) _Float16 sM16[BR][DH];  // mem rows, f16
  __shared__ alignas(16) _Float16 sRH[BR][DH];   // rh rows, f16
  __shared__ alignas(16) float    sG[BR][SGW];   // cols 0:384 Gi | 384:640 Gh (then Ghh)
  __shared__ alignas(16) float    sZ[BR][DH];    // z gate
  __shared__ float sXn[BR], sMn[BR], sRn[BR];

  const int wv   = threadIdx.x >> 5;
  const int lane = threadIdx.x & 31;
  const int R    = blockIdx.x * BR;

  // ---- phase 1: encode (4 rows per wave) ----
#pragma unroll
  for (int i = 0; i < 4; ++i) {
    int lr  = 4 * wv + i;
    int row = R + lr;
    if (row < N) {
      float dt = ts[row] - mem_ts[row];
      float tf[4];
      float ss = 0.f;
#pragma unroll
      for (int t = 0; t < 4; ++t) {
        int j = lane + 32 * t;
        float v = 0.f;
        if (j < DT) v = cosf(dt * time_w[j] + time_b[j]);
        tf[t] = v;
        ss += v * v;
      }
      // expmap0 + proj of time feature
      float un = fmaxf(sqrtf(wsum(ss)), 1e-15f);
      float th = tanhf(un);
      float s  = th / un;
      const float maxnorm = 1.0f - 4e-3f;
      float pn = fmaxf(th, 1e-15f);
      if (pn > maxnorm) s *= maxnorm / pn;

      float xs = 0.f;
#pragma unroll
      for (int t = 0; t < 4; ++t) {
        int j = lane + 32 * t;                 // covers x cols 128..255
        float v = (j < DT) ? tf[t] * s : 0.f;  // zero-pads 228..255
        sXA[lr][DIN + j] = (_Float16)v;
        xs += v * v;
      }
      float mi[4];
      ld4(mem_input + (size_t)row * DIN + 4 * lane, mi);
#pragma unroll
      for (int j = 0; j < 4; ++j) xs += mi[j] * mi[j];
      sth4(&sXA[lr][4 * lane], mi);

      float mr[4];
      ld4(mem + (size_t)row * DH + 4 * lane, mr);
      float ms = 0.f;
#pragma unroll
      for (int j = 0; j < 4; ++j) ms += mr[j] * mr[j];
      sth4(&sM16[lr][4 * lane], mr);

      float xn = fmaxf(sqrtf(wsum(xs)), 1e-15f);
      float mn = fmaxf(sqrtf(wsum(ms)), 1e-15f);
      if (lane == 0) { sXn[lr] = xn; sMn[lr] = mn; }
    } else {            // defined data for the WMMA phases
#pragma unroll
      for (int t = 0; t < 8; ++t) sXA[lr][lane + 32 * t] = (_Float16)0.f;
#pragma unroll
      for (int t = 0; t < 4; ++t) sM16[lr][lane + 32 * t] = (_Float16)0.f;
      if (lane == 0) { sXn[lr] = 1e-15f; sMn[lr] = 1e-15f; }
    }
  }
  __syncthreads();

  // ---- phase 2: 24 Gi + 16 Gh column tiles, 5 per wave.
  //      One B-fragment load feeds BOTH 16-row halves (2 wmma per K-step). ----
#pragma unroll
  for (int it = 0; it < 5; ++it) {
    int t = wv + 8 * it;
    v8f acc0 = {}, acc1 = {};
    if (t < 24) {                               // Gi tile: x[32,256] * Wi[16,256]^T
      const _Float16* B = Wi16 + (size_t)t * 16 * KX;
      __builtin_prefetch(B + (size_t)(lane & 15) * KX, 0, 1);
#pragma unroll
      for (int k = 0; k < KX; k += 32) {
        v16h b  = frag_ld(B, KX, k);
        v16h a0 = frag_ld(&sXA[0][0], KX, k);
        v16h a1 = frag_ld(&sXA[16][0], KX, k);
        acc0 = __builtin_amdgcn_wmma_f32_16x16x32_f16(false, a0, false, b, (short)0, acc0,
                                                      false, false);
        acc1 = __builtin_amdgcn_wmma_f32_16x16x32_f16(false, a1, false, b, (short)0, acc1,
                                                      false, false);
      }
      stacc(&sG[0][0], 0,  t * 16, acc0);
      stacc(&sG[0][0], 16, t * 16, acc1);
    } else {                                    // Gh tile: mem[32,128] * Whrz[16,128]^T
      int tn = t - 24;
      const _Float16* B = Whrz16 + (size_t)tn * 16 * DH;
      __builtin_prefetch(B + (size_t)(lane & 15) * DH, 0, 1);
#pragma unroll
      for (int k = 0; k < DH; k += 32) {
        v16h b  = frag_ld(B, DH, k);
        v16h a0 = frag_ld(&sM16[0][0], DH, k);
        v16h a1 = frag_ld(&sM16[16][0], DH, k);
        acc0 = __builtin_amdgcn_wmma_f32_16x16x32_f16(false, a0, false, b, (short)0, acc0,
                                                      false, false);
        acc1 = __builtin_amdgcn_wmma_f32_16x16x32_f16(false, a1, false, b, (short)0, acc1,
                                                      false, false);
      }
      stacc(&sG[0][0], 0,  GIC + tn * 16, acc0);
      stacc(&sG[0][0], 16, GIC + tn * 16, acc1);
    }
  }
  __syncthreads();

  // ---- phase 3: gates z, r and rh (4 rows per wave) ----
#pragma unroll
  for (int i = 0; i < 4; ++i) {
    int lr  = 4 * wv + i;
    int row = R + lr;
    if (row >= N) continue;
    float xn = sXn[lr], mn = sMn[lr];
    float gh_r[4], gh_z[4], gi_r[4], gi_z[4], m[4], br[4], bz[4];
    ld4(&sG[lr][GIC + 4 * lane], gh_r);            // mem*W_hr^T
    ld4(&sG[lr][GIC + DH + 4 * lane], gh_z);       // mem*W_hz^T
    ld4(&sG[lr][4 * lane], gi_r);                  // x*W_ir^T
    ld4(&sG[lr][2 * DH + 4 * lane], gi_z);         // x*W_iz^T
    ld4(mem + (size_t)row * DH + 4 * lane, m);
    ld4(b_hyp + 0 * DH + 4 * lane, br);
    ld4(b_hyp + 2 * DH + 4 * lane, bz);

    float uh[4], ui[4], t1[4], t2[4], lg[4], z[4], r[4];
    // z gate
    mscale4(gh_z, mn, uh);
    mscale4(gi_z, xn, ui);
    mobius_add4(uh, ui, t1);
    mobius_add4(t1, bz, t2);
    logmap0_4(t2, lg);
#pragma unroll
    for (int j = 0; j < 4; ++j) z[j] = 1.f / (1.f + expf(-lg[j]));
    // r gate
    mscale4(gh_r, mn, uh);
    mscale4(gi_r, xn, ui);
    mobius_add4(uh, ui, t1);
    mobius_add4(t1, br, t2);
    logmap0_4(t2, lg);
#pragma unroll
    for (int j = 0; j < 4; ++j) r[j] = 1.f / (1.f + expf(-lg[j]));

    // rh = mobius_pointwise_mul(r, mem)
    float rm[4], rh[4];
#pragma unroll
    for (int j = 0; j < 4; ++j) rm[j] = r[j] * m[j];
    mscale4(rm, mn, rh);

    st4(&sZ[lr][4 * lane], z);
    sth4(&sRH[lr][4 * lane], rh);
    float rhn = norm4(rh);
    if (lane == 0) sRn[lr] = rhn;
  }
  __syncthreads();

  // ---- phase 4: dependent GEMM  Ghh = rh * W_hh^T  (1 col tile per wave, 2 halves) ----
  {
    const _Float16* B = Whh16 + (size_t)wv * 16 * DH;
    v8f acc0 = {}, acc1 = {};
#pragma unroll
    for (int k = 0; k < DH; k += 32) {
      v16h b  = frag_ld(B, DH, k);
      v16h a0 = frag_ld(&sRH[0][0], DH, k);
      v16h a1 = frag_ld(&sRH[16][0], DH, k);
      acc0 = __builtin_amdgcn_wmma_f32_16x16x32_f16(false, a0, false, b, (short)0, acc0,
                                                    false, false);
      acc1 = __builtin_amdgcn_wmma_f32_16x16x32_f16(false, a1, false, b, (short)0, acc1,
                                                    false, false);
    }
    stacc(&sG[0][0], 0,  GIC + wv * 16, acc0);     // overwrites consumed Gh_r region
    stacc(&sG[0][0], 16, GIC + wv * 16, acc1);
  }
  __syncthreads();

  // ---- phase 5: h_tilde, delta, gated Mobius update, h_new (4 rows per wave) ----
#pragma unroll
  for (int i = 0; i < 4; ++i) {
    int lr  = 4 * wv + i;
    int row = R + lr;
    if (row >= N) continue;

    float g1[4], g2[4], m1[4], m2[4], t1[4], bh[4], ht[4];
    ld4(&sG[lr][GIC + 4 * lane], g1);              // rh*W_hh^T
    ld4(&sG[lr][DH + 4 * lane], g2);               // x*W_ih^T
    mscale4(g1, sRn[lr], m1);
    mscale4(g2, sXn[lr], m2);
    mobius_add4(m1, m2, t1);
    ld4(b_hyp + 1 * DH + 4 * lane, bh);
    mobius_add4(t1, bh, ht);                       // h_tilde

    float m[4], nm[4], dl[4];
    ld4(mem + (size_t)row * DH + 4 * lane, m);
#pragma unroll
    for (int j = 0; j < 4; ++j) nm[j] = -m[j];
    mobius_add4(nm, ht, dl);                       // delta_h

    float z[4], zd_in[4], zd[4], upd[4];
    ld4(&sZ[lr][4 * lane], z);
#pragma unroll
    for (int j = 0; j < 4; ++j) zd_in[j] = z[j] * dl[j];
    float dn = norm4(dl);
    mscale4(zd_in, dn, zd);                        // mobius_pointwise_mul(z, delta)
    mobius_add4(m, zd, upd);                       // updated_memory

    float hv[4], hn[4];
    ld4(hin + (size_t)row * DH + 4 * lane, hv);
    mobius_add4(hv, upd, hn);                      // h_new

    st4(out + (size_t)row * DH + 4 * lane, upd);
    st4(out + (size_t)N * DH + (size_t)row * DH + 4 * lane, hn);
  }
}

// ---------------- host ----------------

extern "C" void kernel_launch(void* const* d_in, const int* in_sizes, int n_in,
                              void* d_out, int out_size, void* d_ws, size_t ws_size,
                              hipStream_t stream) {
  const float* mem_input = (const float*)d_in[0];
  const float* mem       = (const float*)d_in[1];
  const float* ts        = (const float*)d_in[2];
  const float* mem_ts    = (const float*)d_in[3];
  const float* h         = (const float*)d_in[4];
  const float* weight_ih = (const float*)d_in[5];
  const float* weight_hh = (const float*)d_in[6];
  const float* bias      = (const float*)d_in[7];
  const float* time_w    = (const float*)d_in[8];
  const float* time_b    = (const float*)d_in[9];
  const int N = in_sizes[2];
  float* out = (float*)d_out;

  char* ws = (char*)d_ws;
  size_t off = 0;
  auto alloc = [&](size_t bytes) -> char* {
    char* p = ws + off;
    off = (off + bytes + 255) & ~(size_t)255;
    return p;
  };
  _Float16* Wi16   = (_Float16*)alloc((size_t)GIC * KX * sizeof(_Float16));
  _Float16* Whrz16 = (_Float16*)alloc((size_t)GHC * DH * sizeof(_Float16));
  _Float16* Whh16  = (_Float16*)alloc((size_t)DH * DH * sizeof(_Float16));
  float*    b_hyp  = (float*)   alloc(3 * DH * sizeof(float));

  prep_weights<<<dim3((GIC * KX + 255) / 256, 3), 256, 0, stream>>>(
      weight_ih, weight_hh, Wi16, Whrz16, Whh16);
  prep_bias<<<3, 32, 0, stream>>>(bias, b_hyp);

  int blocks = (N + BR - 1) / BR;
  hypgru_fused<<<blocks, 256, 0, stream>>>(mem_input, mem, ts, mem_ts, h,
                                           time_w, time_b,
                                           Wi16, Whrz16, Whh16, b_hyp, out, N);
}
